// Net_44495861186966
// MI455X (gfx1250) — matrix operand order repaired and checked by reading, not compiled, not run on hardware
//
#include <hip/hip_runtime.h>
#include <hip/hip_bf16.h>
#include <cmath>

typedef __attribute__((ext_vector_type(16))) _Float16 v16h;
typedef __attribute__((ext_vector_type(8)))  float    v8f;

#define N_NODES 100000

// ---------------------------------------------------------------- utilities
__global__ void fill_kernel(float* __restrict__ p, float v, long long n) {
    long long i = (long long)blockIdx.x * blockDim.x + threadIdx.x;
    if (i < n) p[i] = v;
}

__global__ void deg_accum_kernel(const long long* __restrict__ dst,
                                 float* __restrict__ deg, int E) {
    int e = blockIdx.x * blockDim.x + threadIdx.x;
    if (e < E) atomicAdd(&deg[(int)dst[e]], 1.0f);
}

__global__ void dinv_kernel(const float* __restrict__ deg,
                            float* __restrict__ dinv, int n) {
    int i = blockIdx.x * blockDim.x + threadIdx.x;
    if (i < n) {
        float d = deg[i];
        dinv[i] = (d > 0.0f) ? rsqrtf(d) : 0.0f;
    }
}

// -------------------------------------------------- weight fragment packing
// B-fragment layout (16-bit B, 32x16 KxN per k-step):
//   lane < 16 : element e = B[K = k0 + (e>>3)*16 + (e&7)    ][N = lane]
//   lane >= 16: element e = B[K = k0 + (e>>3)*16 + 8 + (e&7)][N = lane - 16]
// Packed so each lane's 16 halves are contiguous (32B aligned loads).
__global__ void prep_w1_kernel(const float* __restrict__ W1,
                               _Float16* __restrict__ Bp1) {
    int idx = blockIdx.x * blockDim.x + threadIdx.x;  // 16*4*32*16 = 32768
    if (idx >= 16 * 4 * 32 * 16) return;
    int e    = idx & 15;
    int lane = (idx >> 4) & 31;
    int g    = idx >> 9;        // 0..63
    int nt   = g & 3;
    int kt   = g >> 2;
    int K    = kt * 32 + ((e >> 3) << 4) + ((lane >= 16) ? 8 : 0) + (e & 7);
    int col  = nt * 16 + (lane & 15);
    Bp1[idx] = (_Float16)W1[K * 64 + col];
}

__global__ void prep_w2_kernel(const float* __restrict__ W2,
                               _Float16* __restrict__ Bp2) {
    int idx = blockIdx.x * blockDim.x + threadIdx.x;  // 2*3*32*16 = 3072
    if (idx >= 2 * 3 * 32 * 16) return;
    int e    = idx & 15;
    int lane = (idx >> 4) & 31;
    int g    = idx >> 9;        // 0..5
    int nt   = g % 3;
    int kt   = g / 3;
    int K    = kt * 32 + ((e >> 3) << 4) + ((lane >= 16) ? 8 : 0) + (e & 7);
    int col  = nt * 16 + (lane & 15);
    Bp2[idx] = (col < 40) ? (_Float16)W2[K * 40 + col] : (_Float16)0.0f;
}

// ---------------------------------------------------------------- GEMM 1
// h1[100000,64] = x[100000,512] @ W1[512,64]   (fp16 WMMA, f32 accum)
// block = 128 threads (4 waves); wave w owns N-tile w; grid = 6250 M-tiles.
__global__ void gemm1_kernel(const float* __restrict__ x,
                             const _Float16* __restrict__ Bp1,
                             float* __restrict__ h1) {
    int m0   = blockIdx.x * 16;
    int wave = threadIdx.x >> 5;          // 0..3 -> N tile
    int lane = threadIdx.x & 31;
    int row  = lane & 15;
    int kb   = (lane >> 4) << 3;          // 0 or 8 (A-fragment K base)
    const float* arow = x + (long long)(m0 + row) * 512;

    v8f acc = {};
#pragma unroll
    for (int kt = 0; kt < 16; ++kt) {
        int k0 = kt * 32;
        v16h a;
        const float* p = arow + k0 + kb;
        const float* q = arow + k0 + 16 + kb;
#pragma unroll
        for (int i = 0; i < 8; ++i) a[i] = (_Float16)p[i];
#pragma unroll
        for (int i = 0; i < 8; ++i) a[8 + i] = (_Float16)q[i];

        v16h b = *(const v16h*)(Bp1 + (((kt * 4 + wave) * 32 + lane) << 4));
        acc = __builtin_amdgcn_wmma_f32_16x16x32_f16(
            false, a, false, b, (short)0, acc, false, false);
    }
    // D layout: VGPR r -> M=r (lanes 0-15) / M=8+r (lanes 16-31), N=lane%16
    int col   = wave * 16 + (lane & 15);
    int mbase = (lane < 16) ? 0 : 8;
    float* out = h1 + (long long)m0 * 64 + col;
#pragma unroll
    for (int r = 0; r < 8; ++r)
        out[(long long)(mbase + r) * 64] = acc[r];
}

// ---------------------------------------------------------------- GEMM 2
// h2[100000,40] = a1[100000,64] @ W2[64,40]  (N padded to 48 in fragments)
// block = 96 threads (3 waves); grid = 6250 M-tiles.
__global__ void gemm2_kernel(const float* __restrict__ a1,
                             const _Float16* __restrict__ Bp2,
                             float* __restrict__ h2) {
    int m0   = blockIdx.x * 16;
    int wave = threadIdx.x >> 5;          // 0..2
    int lane = threadIdx.x & 31;
    int row  = lane & 15;
    int kb   = (lane >> 4) << 3;
    const float* arow = a1 + (long long)(m0 + row) * 64;

    v8f acc = {};
#pragma unroll
    for (int kt = 0; kt < 2; ++kt) {
        int k0 = kt * 32;
        v16h a;
        const float* p = arow + k0 + kb;
        const float* q = arow + k0 + 16 + kb;
#pragma unroll
        for (int i = 0; i < 8; ++i) a[i] = (_Float16)p[i];
#pragma unroll
        for (int i = 0; i < 8; ++i) a[8 + i] = (_Float16)q[i];

        v16h b = *(const v16h*)(Bp2 + (((kt * 3 + wave) * 32 + lane) << 4));
        acc = __builtin_amdgcn_wmma_f32_16x16x32_f16(
            false, a, false, b, (short)0, acc, false, false);
    }
    int col   = wave * 16 + (lane & 15);
    int mbase = (lane < 16) ? 0 : 8;
    if (col < 40) {
        float* out = h2 + (long long)m0 * 40 + col;
#pragma unroll
        for (int r = 0; r < 8; ++r)
            out[(long long)(mbase + r) * 40] = acc[r];
    }
}

// ------------------------------------------------------------ aggregation 1
// a1[dst] += h1[src] * dinv[src]*dinv[dst]; 16 float4 lanes per edge.
__global__ void agg1_kernel(const long long* __restrict__ src,
                            const long long* __restrict__ dst,
                            const float* __restrict__ dinv,
                            const float* __restrict__ h1,
                            float* __restrict__ a1, long long total) {
    long long t = (long long)blockIdx.x * blockDim.x + threadIdx.x;
    if (t >= total) return;
    int e  = (int)(t >> 4);
    int c4 = (int)(t & 15);
    int s  = (int)src[e];
    int d  = (int)dst[e];
    float norm = dinv[s] * dinv[d];
    float4 v = ((const float4*)(h1 + (long long)s * 64))[c4];
    float* o = a1 + (long long)d * 64 + (c4 << 2);
    atomicAdd(o + 0, v.x * norm);
    atomicAdd(o + 1, v.y * norm);
    atomicAdd(o + 2, v.z * norm);
    atomicAdd(o + 3, v.w * norm);
}

// self-loop + bias + relu (in place -> becomes layer-2 input)
__global__ void finalize1_kernel(float* __restrict__ a1,
                                 const float* __restrict__ h1,
                                 const float* __restrict__ dinv,
                                 const float* __restrict__ b1, long long total) {
    long long i = (long long)blockIdx.x * blockDim.x + threadIdx.x;
    if (i >= total) return;
    int n = (int)(i >> 6), c = (int)(i & 63);
    float di = dinv[n];
    float v = a1[i] + h1[i] * di * di + b1[c];
    a1[i] = fmaxf(v, 0.0f);
}

// ------------------------------------------------------------ aggregation 2
// out[dst] += h2[src] * norm; 10 float4 lanes per edge (40 channels).
__global__ void agg2_kernel(const long long* __restrict__ src,
                            const long long* __restrict__ dst,
                            const float* __restrict__ dinv,
                            const float* __restrict__ h2,
                            float* __restrict__ out, long long total) {
    long long t = (long long)blockIdx.x * blockDim.x + threadIdx.x;
    if (t >= total) return;
    int e  = (int)(t / 10);
    int c4 = (int)(t - (long long)e * 10);
    int s  = (int)src[e];
    int d  = (int)dst[e];
    float norm = dinv[s] * dinv[d];
    float4 v = ((const float4*)(h2 + (long long)s * 40))[c4];
    float* o = out + (long long)d * 40 + (c4 << 2);
    atomicAdd(o + 0, v.x * norm);
    atomicAdd(o + 1, v.y * norm);
    atomicAdd(o + 2, v.z * norm);
    atomicAdd(o + 3, v.w * norm);
}

// self-loop + bias + log_softmax, one wave32 per node (40 channels)
__global__ void finalize2_kernel(float* __restrict__ out,
                                 const float* __restrict__ h2,
                                 const float* __restrict__ dinv,
                                 const float* __restrict__ b2) {
    int node = blockIdx.x * 8 + (threadIdx.x >> 5);
    int lane = threadIdx.x & 31;
    float di  = dinv[node];
    float di2 = di * di;
    const float* hrow = h2 + (long long)node * 40;
    float* orow = out + (long long)node * 40;

    float v0 = orow[lane] + hrow[lane] * di2 + b2[lane];
    float v1 = (lane < 8)
                 ? (orow[lane + 32] + hrow[lane + 32] * di2 + b2[lane + 32])
                 : -INFINITY;

    float m = fmaxf(v0, v1);
#pragma unroll
    for (int off = 16; off > 0; off >>= 1)
        m = fmaxf(m, __shfl_xor(m, off, 32));

    float s = expf(v0 - m) + ((lane < 8) ? expf(v1 - m) : 0.0f);
#pragma unroll
    for (int off = 16; off > 0; off >>= 1)
        s += __shfl_xor(s, off, 32);

    float lg = logf(s);
    orow[lane] = v0 - m - lg;
    if (lane < 8) orow[lane + 32] = v1 - m - lg;
}

// ---------------------------------------------------------------- launcher
extern "C" void kernel_launch(void* const* d_in, const int* in_sizes, int n_in,
                              void* d_out, int out_size, void* d_ws, size_t ws_size,
                              hipStream_t stream) {
    const float*     x    = (const float*)d_in[0];
    const long long* eidx = (const long long*)d_in[1];
    const float*     W1   = (const float*)d_in[2];
    const float*     b1   = (const float*)d_in[3];
    const float*     W2   = (const float*)d_in[4];
    const float*     b2   = (const float*)d_in[5];
    float* out = (float*)d_out;

    const int N = N_NODES;
    const int E = in_sizes[1] / 2;
    const long long* src = eidx;
    const long long* dst = eidx + E;

    // workspace carve-up (all offsets keep 32B alignment)
    float* deg  = (float*)d_ws;                 // 100000  (padded slot 100032)
    float* dinv = deg  + 100032;
    float* h1   = dinv + 100032;                // 100000*64
    float* a1   = h1   + (long long)N * 64;     // 100000*64
    float* h2   = a1   + (long long)N * 64;     // 100000*40
    _Float16* Bp1 = (_Float16*)(h2 + (long long)N * 40);   // 32768 halves
    _Float16* Bp2 = Bp1 + 32768;                            // 3072 halves

    const int B = 256;

    // degree (self-loop contributes 1) and normalization
    fill_kernel<<<(N + B - 1) / B, B, 0, stream>>>(deg, 1.0f, N);
    deg_accum_kernel<<<(E + B - 1) / B, B, 0, stream>>>(dst, deg, E);
    dinv_kernel<<<(N + B - 1) / B, B, 0, stream>>>(deg, dinv, N);

    // zero accumulators (d_out is poisoned by harness; must re-zero every call)
    long long n1 = (long long)N * 64;
    long long n2 = (long long)N * 40;
    fill_kernel<<<(unsigned)((n1 + B - 1) / B), B, 0, stream>>>(a1, 0.0f, n1);
    fill_kernel<<<(unsigned)((n2 + B - 1) / B), B, 0, stream>>>(out, 0.0f, n2);

    // weight fragment packing
    prep_w1_kernel<<<32768 / B, B, 0, stream>>>(W1, Bp1);
    prep_w2_kernel<<<(3072 + B - 1) / B, B, 0, stream>>>(W2, Bp2);

    // layer 1
    gemm1_kernel<<<N / 16, 128, 0, stream>>>(x, Bp1, h1);
    long long tot1 = (long long)E * 16;
    agg1_kernel<<<(unsigned)((tot1 + B - 1) / B), B, 0, stream>>>(src, dst, dinv, h1, a1, tot1);
    finalize1_kernel<<<(unsigned)((n1 + B - 1) / B), B, 0, stream>>>(a1, h1, dinv, b1, n1);

    // layer 2
    gemm2_kernel<<<N / 16, 96, 0, stream>>>(a1, Bp2, h2);
    long long tot2 = (long long)E * 10;
    agg2_kernel<<<(unsigned)((tot2 + B - 1) / B), B, 0, stream>>>(src, dst, dinv, h2, out, tot2);
    finalize2_kernel<<<N / 8, 256, 0, stream>>>(out, h2, dinv, b2);
}
